// Net_49563922596719
// MI455X (gfx1250) — matrix operand order repaired and checked by reading, not compiled, not run on hardware
//
#include <hip/hip_runtime.h>
#include <hip/hip_bf16.h>
#include <math.h>

// ---------------------------------------------------------------------------
// Problem constants (reference: B=8, C=64, H=256, W=256, rfft width 129)
// ---------------------------------------------------------------------------
#define BD   8
#define CD   64
#define HD   256
#define WD   256
#define WF   129
#define PSP  (HD * WD)          // 65536 spatial pixels
#define PFR  (HD * WF)          // 33024 freq bins per (b,c)
static const size_t S_ELEMS = (size_t)BD * CD * PSP;   // 33,554,432
static const size_t F_ELEMS = (size_t)BD * CD * PFR;   // 16,908,288

// Use the CDNA5 async global->LDS DMA path (inline asm, ASYNCcnt-tracked).
#define USE_ASYNC_LDS 1

// ---------------------------------------------------------------------------
// CDNA5 WMMA types
// ---------------------------------------------------------------------------
typedef __attribute__((ext_vector_type(16))) __bf16 v16bf;
typedef __attribute__((ext_vector_type(8)))  __bf16 v8bf;
typedef __attribute__((ext_vector_type(8)))  float  v8f;

__device__ __forceinline__ float2 cplx_mul(float2 a, float2 b) {
    return make_float2(a.x * b.x - a.y * b.y, a.x * b.y + a.y * b.x);
}

#if USE_ASYNC_LDS
// 32-bit LDS byte offset of a generic pointer into shared memory
__device__ __forceinline__ unsigned lds_byte_off(const void* p) {
    return (unsigned)(unsigned long long)(const __attribute__((address_space(3))) char*)p;
}
#endif

// ---------------------------------------------------------------------------
// 1x1 conv as WMMA GEMM:  out[b, m, n] = sum_k W[m,k] * X[b,k,n] + bias[m]
// X supplied as two pointers split at channel Csplit (for the phase-concat
// conv); for ordinary convs pass inB == inA and Csplit == Cin.
//
// Tile: 64(M) x 32(N) per 256-thread block; K stepped by 32 through LDS in
// bf16. LDS layouts are arranged so each wave fetches its WMMA fragments with
// ds_load_b128 (A: row-major K-runs of 8; B: stored transposed so the K-run
// of 8 (two per lane) is contiguous). Rows padded to 40 elems (80B, 16B-aligned).
// ---------------------------------------------------------------------------
#define LDA 40
__global__ __launch_bounds__(256)
void k_conv1x1_wmma(const float* __restrict__ inA, const float* __restrict__ inB,
                    int Cin, int Csplit, int P,
                    const float* __restrict__ Wm, const float* __restrict__ bias,
                    float* __restrict__ out, int doLrelu, float addConst)
{
    __shared__ __bf16 lA [64 * LDA];  // [m][k]  (k in 0..31 used)
    __shared__ __bf16 lBt[32 * LDA];  // [n][k]  transposed activations

    const unsigned b      = blockIdx.y;
    const unsigned n_base = blockIdx.x * 32u;
    const unsigned tid    = threadIdx.x;
    const unsigned lane   = tid & 31u;
    const unsigned wave   = tid >> 5;
    const unsigned hi     = (lane >> 4) & 1u;
    const unsigned l16    = lane & 15u;
    const unsigned m0     = (wave & 3u) * 16u;   // 4 M-tiles cover M=64
    const unsigned n0     = (wave >> 2) * 16u;   // 2 N-tiles cover N=32

    const unsigned sn  = tid & 31u;   // staging: fixed n per thread
    const unsigned skk = tid >> 5;    // staging: base k per thread (+8 per j)

    v8f acc = {};

    for (int k0 = 0; k0 < Cin; k0 += 32) {
        // ---- stage A (weights), coalesced fp32 -> bf16 ----
        #pragma unroll
        for (unsigned j = 0; j < 8; ++j) {
            unsigned li = tid + 256u * j;          // 2048 elements
            unsigned m  = li >> 5;
            unsigned kk = li & 31u;
            lA[m * LDA + kk] = (__bf16)Wm[m * (unsigned)Cin + (unsigned)k0 + kk];
        }
        // ---- stage B (activations) transposed: lBt[n][k] = X[k][n] ----
        #pragma unroll
        for (unsigned j = 0; j < 4; ++j) {
            unsigned kk = skk + 8u * j;
            unsigned k  = (unsigned)k0 + kk;
            const float* src;
            unsigned chan;
            if ((int)k < Csplit) { src = inA; chan = b * (unsigned)Csplit + k; }
            else                 { src = inB; chan = b * (unsigned)(Cin - Csplit) + (k - (unsigned)Csplit); }
            lBt[sn * LDA + kk] = (__bf16)src[chan * (unsigned)P + n_base + sn];
        }
        if (k0 + 32 < Cin) {
            // hint next weight K-slice into cache (global_prefetch_b8)
            __builtin_prefetch(Wm + (size_t)(m0 + l16) * Cin + (k0 + 32), 0, 1);
        }
        __syncthreads();

        // ---- fragment loads: 2x ds_load_b128 each (ISA 7.12.2 layouts) ----
        const v8bf a_lo = *(const v8bf*)&lA [(m0 + l16) * LDA +  8u * hi];
        const v8bf a_hi = *(const v8bf*)&lA [(m0 + l16) * LDA + 16u + 8u * hi];
        const v8bf b_lo = *(const v8bf*)&lBt[(n0 + l16) * LDA +  8u * hi];
        const v8bf b_hi = *(const v8bf*)&lBt[(n0 + l16) * LDA + 16u + 8u * hi];
        v16bf afrag = __builtin_shufflevector(a_lo, a_hi, 0, 1, 2, 3, 4, 5, 6, 7,
                                              8, 9, 10, 11, 12, 13, 14, 15);
        v16bf bfrag = __builtin_shufflevector(b_lo, b_hi, 0, 1, 2, 3, 4, 5, 6, 7,
                                              8, 9, 10, 11, 12, 13, 14, 15);
        acc = __builtin_amdgcn_wmma_f32_16x16x32_bf16(
                  /*neg_a=*/false, afrag, /*neg_b=*/false, bfrag,
                  /*c_mod=*/(short)0, acc, /*reuse_a=*/false, /*reuse_b=*/false);
        __syncthreads();
    }

    // ---- epilogue: C/D layout VGPR r -> M = m0 + r + 8*hi, N = n0 + l16 ----
    const unsigned n = n_base + n0 + l16;
    #pragma unroll
    for (unsigned r = 0; r < 8; ++r) {
        unsigned m = m0 + r + 8u * hi;
        float v = acc[r];
        if (bias) v += bias[m];
        if (doLrelu) v = (v > 0.0f) ? v : 0.1f * v;
        v += addConst;
        out[(b * (unsigned)CD + m) * (unsigned)P + n] = v;
    }
}

// ---------------------------------------------------------------------------
// 256-point radix-2 Stockham FFT in LDS. 128 threads, one butterfly each per
// stage; natural-order output; returns buffer holding the result.
// sign = -1 forward, +1 inverse (no 1/N scaling here).
// ---------------------------------------------------------------------------
__device__ float2* fft256_lds(float2* s0, float2* s1, float sign)
{
    float2* src = s0;
    float2* dst = s1;
    const int t = threadIdx.x;            // 0..127
    for (int Ns = 1; Ns < 256; Ns <<= 1) {
        __syncthreads();
        float2 a = src[t];
        float2 b = src[t + 128];
        int   j  = t & (Ns - 1);
        float ang = sign * 3.14159265358979323846f * (float)j / (float)Ns;
        float sn, cs;
        __sincosf(ang, &sn, &cs);
        float2 bw = cplx_mul(b, make_float2(cs, sn));
        int base = ((t - j) << 1) + j;    // (t/Ns)*2Ns + j
        dst[base]      = make_float2(a.x + bw.x, a.y + bw.y);
        dst[base + Ns] = make_float2(a.x - bw.x, a.y - bw.y);
        float2* tmp = src; src = dst; dst = tmp;
    }
    __syncthreads();
    return src;
}

// rows of real input -> half-spectrum (129 bins) per row
__global__ __launch_bounds__(128)
void k_fft_row_fwd(const float* __restrict__ x, float2* __restrict__ out)
{
    __shared__ float  sraw[256];
    __shared__ float2 s0[256], s1[256];
    const size_t row = blockIdx.x;
    const int t = threadIdx.x;
    const float* xr = x + row * (size_t)WD;
#if USE_ASYNC_LDS
    {
        // per-lane 8B global->LDS DMA (ASYNCcnt), then drain + barrier
        unsigned           ldst = lds_byte_off(&sraw[t * 2]);
        unsigned long long ga   = (unsigned long long)(xr + t * 2);
        asm volatile("global_load_async_to_lds_b64 %0, %1, off"
                     :: "v"(ldst), "v"(ga) : "memory");
        asm volatile("s_wait_asynccnt 0" ::: "memory");
    }
    __syncthreads();
    s0[t]       = make_float2(sraw[t], 0.0f);
    s0[t + 128] = make_float2(sraw[t + 128], 0.0f);
#else
    s0[t]       = make_float2(xr[t], 0.0f);
    s0[t + 128] = make_float2(xr[t + 128], 0.0f);
#endif
    float2* res = fft256_lds(s0, s1, -1.0f);
    float2* o = out + row * (size_t)WF;
    o[t] = res[t];
    if (t == 0) o[128] = res[128];
}

// in-place column FFT across H for every (image, wf) column; sign selectable
__global__ __launch_bounds__(128)
void k_fft_col(float2* __restrict__ data, float sign)
{
    __shared__ float2 s0[256], s1[256];
    const int img = blockIdx.x / WF;      // b*C + c
    const int wf  = blockIdx.x % WF;
    const int t   = threadIdx.x;
    float2* basep = data + (size_t)img * HD * WF + wf;
    s0[t]       = basep[(size_t)t * WF];
    s0[t + 128] = basep[(size_t)(t + 128) * WF];
    float2* res = fft256_lds(s0, s1, sign);
    basep[(size_t)t * WF]         = res[t];
    basep[(size_t)(t + 128) * WF] = res[t + 128];
}

// half-spectrum row -> Hermitian-expanded inverse FFT -> |real| * 1/(H*W)
__global__ __launch_bounds__(128)
void k_fft_row_inv_abs(const float2* __restrict__ spec, float* __restrict__ out)
{
    __shared__ float2 s0[256], s1[256];
    const size_t row = blockIdx.x;
    const int t = threadIdx.x;
    const float2* sp = spec + row * (size_t)WF;
    float2 v = sp[t];
    s0[t] = v;
    if (t == 0) s0[128] = sp[128];
    if (t >= 1) s0[256 - t] = make_float2(v.x, -v.y);   // conj symmetry
    float2* res = fft256_lds(s0, s1, +1.0f);
    const float scale = 1.0f / ((float)HD * (float)WD);
    float* o = out + row * (size_t)WD;
    o[t]       = fabsf(res[t].x * scale);
    o[t + 128] = fabsf(res[t + 128].x * scale);
}

// ---------------------------------------------------------------------------
// Elementwise / small kernels
// ---------------------------------------------------------------------------
__global__ __launch_bounds__(256)
void k_amp_pha(const float2* __restrict__ c, float* __restrict__ amp,
               float* __restrict__ pha, size_t n)
{
    size_t i = (size_t)blockIdx.x * 256 + threadIdx.x;
    if (i < n) {
        float2 v = c[i];
        amp[i] = sqrtf(v.x * v.x + v.y * v.y);
        pha[i] = atan2f(v.y, v.x);
    }
}

__global__ __launch_bounds__(256)
void k_gap(const float* __restrict__ a, const float* __restrict__ b,
           float* __restrict__ gap)
{
    __shared__ float red[256];
    const int bc = blockIdx.x;
    const float* pa = a + (size_t)bc * PFR;
    const float* pb = b + (size_t)bc * PFR;
    float s = 0.0f;
    for (int i = threadIdx.x; i < PFR; i += 256) s += pa[i] + pb[i];
    red[threadIdx.x] = s;
    __syncthreads();
    for (int off = 128; off > 0; off >>= 1) {
        if (threadIdx.x < off) red[threadIdx.x] += red[threadIdx.x + off];
        __syncthreads();
    }
    if (threadIdx.x == 0) gap[bc] = red[0] / (float)PFR;
}

__global__ __launch_bounds__(64)
void k_attn(const float* __restrict__ gap, const float* __restrict__ fc_w,
            const float* __restrict__ fcs0_w, const float* __restrict__ fcs0_b,
            const float* __restrict__ fcs1_w, const float* __restrict__ fcs1_b,
            float* __restrict__ attn)
{
    __shared__ float sg[64];
    __shared__ float z[8];
    const int b = blockIdx.x, c = threadIdx.x;
    sg[c] = gap[b * CD + c];
    __syncthreads();
    if (c < 8) {
        float acc = 0.0f;
        for (int k = 0; k < 64; ++k) acc += fc_w[c * 64 + k] * sg[k];
        z[c] = fmaxf(acc, 0.0f);
    }
    __syncthreads();
    float a0 = fcs0_b[c], a1 = fcs1_b[c];
    for (int j = 0; j < 8; ++j) {
        a0 += fcs0_w[c * 8 + j] * z[j];
        a1 += fcs1_w[c * 8 + j] * z[j];
    }
    float m = fmaxf(a0, a1);
    float e0 = __expf(a0 - m), e1 = __expf(a1 - m);
    // softmax[1] + 0.5, then sliced attn + 0.5  ->  softmax1 + 1.0
    attn[b * CD + c] = e1 / (e0 + e1) + 1.0f;
}

__global__ __launch_bounds__(256)
void k_feats_v(const float* __restrict__ ampA, const float* __restrict__ ampB,
               const float* __restrict__ attn, float* __restrict__ V, size_t n)
{
    size_t i = (size_t)blockIdx.x * 256 + threadIdx.x;
    if (i < n) {
        int bc = (int)(i / (size_t)PFR);
        V[i] = (ampA[i] + ampB[i]) * attn[bc];
    }
}

__global__ __launch_bounds__(256)
void k_spec(const float* __restrict__ amp, const float* __restrict__ pha,
            float2* __restrict__ spec, size_t n)
{
    size_t i = (size_t)blockIdx.x * 256 + threadIdx.x;
    if (i < n) {
        float a = amp[i], p = pha[i];
        float sn, cs;
        __sincosf(p, &sn, &cs);
        // real = a*cos + EPS, imag = a*sin + EPS, spec = complex + EPS
        spec[i] = make_float2(a * cs + 2e-8f, a * sn + 1e-8f);
    }
}

// ---------------------------------------------------------------------------
// Host-side orchestration
// ---------------------------------------------------------------------------
extern "C" void kernel_launch(void* const* d_in, const int* in_sizes, int n_in,
                              void* d_out, int out_size, void* d_ws, size_t ws_size,
                              hipStream_t stream)
{
    (void)in_sizes; (void)n_in; (void)out_size; (void)ws_size;

    const float* t2f    = (const float*)d_in[0];
    const float* t1f    = (const float*)d_in[1];
    const float* pre1_w = (const float*)d_in[2];
    const float* pre1_b = (const float*)d_in[3];
    const float* pre2_w = (const float*)d_in[4];
    const float* pre2_b = (const float*)d_in[5];
    const float* amp1_w = (const float*)d_in[6];
    const float* amp1_b = (const float*)d_in[7];
    const float* amp2_w = (const float*)d_in[8];
    const float* amp2_b = (const float*)d_in[9];
    const float* pha1_w = (const float*)d_in[10];
    const float* pha1_b = (const float*)d_in[11];
    const float* pha2_w = (const float*)d_in[12];
    const float* pha2_b = (const float*)d_in[13];
    const float* post_w = (const float*)d_in[14];
    const float* post_b = (const float*)d_in[15];
    const float* fc_w   = (const float*)d_in[16];
    const float* fcs0_w = (const float*)d_in[17];
    const float* fcs0_b = (const float*)d_in[18];
    const float* fcs1_w = (const float*)d_in[19];
    const float* fcs1_b = (const float*)d_in[20];

    // workspace carve-up (floats): S + 2F + 2F + 4F + small
    float*  ws   = (float*)d_ws;
    float*  spat = ws;                                   // S floats
    float2* c1   = (float2*)(ws + S_ELEMS);              // F complex
    float2* c2   = (float2*)(ws + S_ELEMS + 2 * F_ELEMS);// F complex
    float*  ampA = ws + S_ELEMS + 4 * F_ELEMS;
    float*  phaA = ampA + F_ELEMS;
    float*  ampB = phaA + F_ELEMS;
    float*  phaB = ampB + F_ELEMS;
    float*  gap  = phaB + F_ELEMS;                       // B*C
    float*  attn = gap + BD * CD;                        // B*C

    float*  fV   = (float*)c1;                           // reuse: feats_V / pha_mid
    float*  fmid = fV + F_ELEMS;                         // reuse: amp_mid

    const dim3 blk256(256), blk128(128), blk64(64);
    const dim3 gConvSp(PSP / 32, BD);                    // spatial GEMM tiles
    const dim3 gConvFr(PFR / 32, BD);                    // freq GEMM tiles
    const int  gRows = BD * CD * HD;                     // 131072 row-FFT blocks
    const int  gCols = BD * CD * WF;                     // 66048 col-FFT blocks
    const int  gEltF = (int)((F_ELEMS + 255) / 256);

    // ---- stream A: pre1 conv (+EPS) -> rfft2 -> amp/pha ----
    k_conv1x1_wmma<<<gConvSp, blk256, 0, stream>>>(t2f, t2f, CD, CD, PSP,
                                                   pre1_w, pre1_b, spat, 0, 1e-8f);
    k_fft_row_fwd<<<gRows, blk128, 0, stream>>>(spat, c1);
    k_fft_col<<<gCols, blk128, 0, stream>>>(c1, -1.0f);
    k_amp_pha<<<gEltF, blk256, 0, stream>>>(c1, ampA, phaA, F_ELEMS);

    // ---- stream B: pre2 conv (+EPS) -> rfft2 -> amp/pha ----
    k_conv1x1_wmma<<<gConvSp, blk256, 0, stream>>>(t1f, t1f, CD, CD, PSP,
                                                   pre2_w, pre2_b, spat, 0, 1e-8f);
    k_fft_row_fwd<<<gRows, blk128, 0, stream>>>(spat, c2);
    k_fft_col<<<gCols, blk128, 0, stream>>>(c2, -1.0f);
    k_amp_pha<<<gEltF, blk256, 0, stream>>>(c2, ampB, phaB, F_ELEMS);

    // ---- SK attention on amplitudes ----
    k_gap<<<BD * CD, blk256, 0, stream>>>(ampA, ampB, gap);
    k_attn<<<BD, blk64, 0, stream>>>(gap, fc_w, fcs0_w, fcs0_b, fcs1_w, fcs1_b, attn);
    k_feats_v<<<gEltF, blk256, 0, stream>>>(ampA, ampB, attn, fV, F_ELEMS);

    // ---- amplitude branch: conv -> lrelu -> conv ----
    k_conv1x1_wmma<<<gConvFr, blk256, 0, stream>>>(fV, fV, CD, CD, PFR,
                                                   amp1_w, amp1_b, fmid, 1, 0.0f);
    k_conv1x1_wmma<<<gConvFr, blk256, 0, stream>>>(fmid, fmid, CD, CD, PFR,
                                                   amp2_w, amp2_b, ampA, 0, 0.0f);

    // ---- phase branch: concat(2C) conv -> lrelu -> conv ----
    k_conv1x1_wmma<<<gConvFr, blk256, 0, stream>>>(phaA, phaB, 2 * CD, CD, PFR,
                                                   pha1_w, pha1_b, fV, 1, 0.0f);
    k_conv1x1_wmma<<<gConvFr, blk256, 0, stream>>>(fV, fV, CD, CD, PFR,
                                                   pha2_w, pha2_b, ampB, 0, 0.0f);

    // ---- rebuild spectrum, irfft2, |.|, post conv ----
    k_spec<<<gEltF, blk256, 0, stream>>>(ampA, ampB, c2, F_ELEMS);
    k_fft_col<<<gCols, blk128, 0, stream>>>(c2, +1.0f);
    k_fft_row_inv_abs<<<gRows, blk128, 0, stream>>>(c2, spat);
    k_conv1x1_wmma<<<gConvSp, blk256, 0, stream>>>(spat, spat, CD, CD, PSP,
                                                   post_w, post_b, (float*)d_out, 0, 0.0f);
}
